// SignatureComputer_48473000902863
// MI455X (gfx1250) — matrix operand order repaired and checked by reading, not compiled, not run on hardware
//
#include <hip/hip_runtime.h>

typedef __attribute__((ext_vector_type(16))) _Float16 v16h;
typedef __attribute__((ext_vector_type(8)))  _Float16 v8h;
typedef __attribute__((ext_vector_type(8)))  float    v8f;
typedef __attribute__((ext_vector_type(4)))  float    v4f;

#define SEQ      256
#define WIN      32
#define KDIM     32      // padded K (31 real time steps)
#define GDIM     585     // 1 + 8 + 64 + 512
#define GPAD     592     // 37 * 16
#define NTILES   37
#define TOTAL    4680    // 8 + 64 + 512 + 4096
#define LN_EPS   1e-5f

union HV16 { v16h v; struct { v8h lo, hi; } h; };

// g -> G[t][g] value; with compile-time g (fully unrolled callers) all array
// indexing is static and the branches fold away.
__device__ __forceinline__ float gfun(int g, const float p[8]) {
    const float s3 = 1.0f / 31.0f;
    const float s4 = s3 * s3;
    if (g == 0)   return 1.0f;                                     // lvl1
    if (g < 9)    return p[g - 1];                                 // lvl2
    if (g < 73) { int q = g - 9;  return p[q >> 3] * p[q & 7] * s3; }          // lvl3
    if (g < GDIM) { int q = g - 73;
                    return p[(q >> 6)] * p[(q >> 3) & 7] * p[q & 7] * s4; }    // lvl4
    return 0.0f;                                                   // N pad
}

__global__ __launch_bounds__(32) void sig_wmma_kernel(
    const float* __restrict__ features,   // [B, 256, 7]
    const float* __restrict__ gamma,      // [4680]
    const float* __restrict__ beta,       // [4680]
    const int*   __restrict__ t_ptr,      // scalar t
    float*       __restrict__ out)        // [B, 4680]
{
    __shared__ __align__(16) float     path[WIN * 8];        // [s][d]   1 KB
    __shared__ __align__(16) _Float16  incT[16 * KDIM];      // [m][k]   1 KB
    __shared__ __align__(16) _Float16  gtile[16 * KDIM];     // [n][k]   1 KB
    __shared__ __align__(16) float     S[GPAD * 8];          // [g][m]  18.5 KB

    const int lane = threadIdx.x;   // 0..31 ; doubles as time index t
    const int b    = blockIdx.x;
    const int t    = *t_ptr;

    // ---- 1. load this lane's path point (front-padded window row) ----
    int src = t - (WIN - 1) + lane;
    if (src < 0) src = 0;
    float p[8];
    p[0] = (float)lane * (1.0f / 31.0f);                    // time channel
    const float* frow = features + ((size_t)b * SEQ + src) * 7;
    #pragma unroll
    for (int d = 0; d < 7; ++d) p[d + 1] = frow[d];

    #pragma unroll
    for (int d = 0; d < 8; ++d) path[lane * 8 + d] = p[d];
    asm volatile("s_wait_dscnt 0" ::: "memory");

    // ---- 2. inc[t][i] -> A matrix (incT); rows 8..15 and k=31 are zero ----
    #pragma unroll
    for (int i = 0; i < 8; ++i) {
        float inc = 0.0f;
        if (lane < 31) inc = path[(lane + 1) * 8 + i] - p[i];
        incT[i * KDIM + lane] = (_Float16)inc;
    }
    #pragma unroll
    for (int i = 8; i < 16; ++i) incT[i * KDIM + lane] = (_Float16)0.0f;
    asm volatile("s_wait_dscnt 0" ::: "memory");

    // ---- 3. A fragment (16x32 f16): lane<16 K{0-7,16-23}, lane>=16 K{8-15,24-31}
    const int m  = lane & 15;
    const int kh = (lane >> 4) * 8;       // A half-offset
    HV16 ua;
    ua.h.lo = *(const v8h*)&incT[m * KDIM + kh];
    ua.h.hi = *(const v8h*)&incT[m * KDIM + kh + 16];

    const int kc = (lane >> 4) * 16;      // unused (0 or 16); B loads use kh like A

    // ---- 4. fully-unrolled tile loop: build 16-col G slice, WMMA, stage ----
    float sum = 0.0f, sq = 0.0f;
    #pragma unroll
    for (int nt = 0; nt < NTILES; ++nt) {
        // In-place launder: zero-cost opaque redefinition of p[] per tile so
        // IR-level CSE cannot merge identical products across tiles (keeps
        // live ranges tile-local, VGPR count < 256) without emitting copies.
        #pragma unroll
        for (int d = 0; d < 8; ++d) asm volatile("" : "+v"(p[d]));

        // build G^T slice for columns [16*nt, 16*nt+16) — g is compile-time
        #pragma unroll
        for (int i = 0; i < 16; ++i)
            gtile[i * KDIM + lane] = (_Float16)gfun(nt * 16 + i, p);
        asm volatile("s_wait_dscnt 0" ::: "memory");

        HV16 ub;
        ub.h.lo = *(const v8h*)&gtile[m * KDIM + kh];
        ub.h.hi = *(const v8h*)&gtile[m * KDIM + kh + 16];

        v8f c = {};
        c = __builtin_amdgcn_wmma_f32_16x16x32_f16(
                false, ua.v, false, ub.v, (short)0, c, false, false);

        #pragma unroll
        for (int r = 0; r < 8; ++r) { sum += c[r]; sq += c[r] * c[r]; }

        if (lane < 16) {                  // rows M=0..7 live in lanes 0..15
            // column-major stage: 8 consecutive VGPRs -> two b128 stores
            v4f* dst = (v4f*)&S[(nt * 16 + lane) * 8];
            dst[0] = *(const v4f*)&c;
            dst[1] = *((const v4f*)&c + 1);
        }
        // same-wave DS is in-order: next tile's gtile stores cannot bypass
        // this tile's fragment loads.
    }

    // ---- 5. wave32 reduction for LayerNorm stats (pads contribute exact 0)
    #pragma unroll
    for (int off = 16; off > 0; off >>= 1) {
        sum += __shfl_xor(sum, off, 32);
        sq  += __shfl_xor(sq,  off, 32);
    }
    const float mean = sum * (1.0f / (float)TOTAL);
    const float var  = sq  * (1.0f / (float)TOTAL) - mean * mean;
    const float rs   = rsqrtf(var + LN_EPS);

    asm volatile("s_wait_dscnt 0" ::: "memory");

    // ---- 6. normalize + scatter, vectorized by 4 (levels/rows align to 4) ----
    float* outb = out + (size_t)b * TOTAL;
    for (int it = 0; it < (TOTAL / 128) + 1; ++it) {
        const int gi = lane + it * 32;            // group index, 4 elems each
        if (gi >= TOTAL / 4) break;
        const int o0 = gi * 4;

        int base, stride;
        if (o0 < 8)        { base = o0;                                 stride = 1; }
        else if (o0 < 72)  { int q = o0 - 8;   base = (1  + (q & 7))  * 8 + (q >> 3); stride = 8; }
        else if (o0 < 584) { int q = o0 - 72;  base = (9  + (q & 63)) * 8 + (q >> 6); stride = 8; }
        else               { int q = o0 - 584; base = (73 + (q & 511))* 8 + (q >> 9); stride = 8; }

        const v4f gm = *(const v4f*)&gamma[o0];
        const v4f bt = *(const v4f*)&beta[o0];
        v4f r;
        r[0] = (S[base             ] - mean) * rs * gm[0] + bt[0];
        r[1] = (S[base +     stride] - mean) * rs * gm[1] + bt[1];
        r[2] = (S[base + 2 * stride] - mean) * rs * gm[2] + bt[2];
        r[3] = (S[base + 3 * stride] - mean) * rs * gm[3] + bt[3];
        *(v4f*)&outb[o0] = r;
    }
}

extern "C" void kernel_launch(void* const* d_in, const int* in_sizes, int n_in,
                              void* d_out, int out_size, void* d_ws, size_t ws_size,
                              hipStream_t stream) {
    const float* features = (const float*)d_in[0];
    const float* gamma    = (const float*)d_in[1];
    const float* beta     = (const float*)d_in[2];
    const int*   t        = (const int*)d_in[3];
    float*       out      = (float*)d_out;

    const int batches = in_sizes[0] / (SEQ * 7);   // 8192
    sig_wmma_kernel<<<batches, 32, 0, stream>>>(features, gamma, beta, t, out);
}
